// SelfAttentionModel_39848706572694
// MI455X (gfx1250) — matrix operand order repaired and checked by reading, not compiled, not run on hardware
//
#include <hip/hip_runtime.h>

typedef _Float16 f16;
typedef __attribute__((ext_vector_type(16))) _Float16 v16h;
typedef __attribute__((ext_vector_type(8)))  _Float16 v8h;
typedef __attribute__((ext_vector_type(4)))  _Float16 v4h;
typedef __attribute__((ext_vector_type(8)))  float    v8f;
typedef __attribute__((ext_vector_type(4)))  float    v4f;

#define BATCH 2
#define SEQ   2048
#define HEADS 16
#define DH    64

#define BQ   64      // queries per block (4 waves x 16)
#define BK   64      // keys per chunk
#define KSTR 72      // kbuf row stride in halves (144B rows, 16B aligned)
#define VSTR 72      // vbufT row stride in halves (144B rows, 16B aligned)
#define PSTR 72      // pbuf row stride in halves (144B rows, 16B aligned)

static __device__ __forceinline__ v8f wmma_f16(v16h a, v16h b, v8f c) {
  // D(16x16 f32) = A(16x32 f16) * B(32x16 f16) + C
  return __builtin_amdgcn_wmma_f32_16x16x32_f16(
      /*neg_a=*/false, a, /*neg_b=*/false, b,
      /*c_mod=*/(short)0, c, /*reuse_a=*/false, /*reuse_b=*/false);
}

// DPP cross-lane mov (VALU, no LDS): butterfly within each 16-lane group.
template <int CTRL>
static __device__ __forceinline__ float dpp_bfly(float x) {
  return __int_as_float(__builtin_amdgcn_update_dpp(
      0, __float_as_int(x), CTRL, /*row_mask=*/0xF, /*bank_mask=*/0xF,
      /*bound_ctrl=*/true));
}
// quad_perm [1,0,3,2]=0xB1, [2,3,0,1]=0x4E, row_half_mirror=0x141, row_mirror=0x140
static __device__ __forceinline__ float redmax16(float x) {
  x = fmaxf(x, dpp_bfly<0xB1>(x));
  x = fmaxf(x, dpp_bfly<0x4E>(x));
  x = fmaxf(x, dpp_bfly<0x141>(x));
  x = fmaxf(x, dpp_bfly<0x140>(x));
  return x;
}
static __device__ __forceinline__ float redsum16(float x) {
  x += dpp_bfly<0xB1>(x);
  x += dpp_bfly<0x4E>(x);
  x += dpp_bfly<0x141>(x);
  x += dpp_bfly<0x140>(x);
  return x;
}

__global__ __launch_bounds__(128)
void flash_attn_f16_wmma(const float* __restrict__ Q,
                         const float* __restrict__ K,
                         const float* __restrict__ V,
                         const float* __restrict__ Mask,
                         const int*   __restrict__ InvScale,
                         float* __restrict__ Out) {
  __shared__ f16 kbuf[BK * KSTR];        // K chunk, row-major [key][d]
  __shared__ f16 vbufT[DH * VSTR];       // V chunk, transposed [d][key]
  __shared__ f16 pbuf[4 * 16 * PSTR];    // per-wave P tile [row][key]

  const int tid  = threadIdx.x;
  const int wave = tid >> 5;
  const int lane = tid & 31;
  const int n16  = lane & 15;            // column / row-in-A index
  const int hi   = lane >> 4;            // lane-half select

  const int bh = blockIdx.y;             // b*HEADS + h
  const int b  = bh / HEADS;
  const int h  = bh % HEADS;
  const int q0 = blockIdx.x * BQ + wave * 16;

  const float LOG2E  = 1.4426950408889634f;
  const float kscale = LOG2E / (float)InvScale[0]; // log2-domain score scale

  // ---- load Q tile (16 rows) into A-layout fragments, fp32 -> f16 ----
  union AF { v16h v; f16 e[16]; };
  v16h qa[2];
  {
    const int qrow = q0 + n16;           // lane holds row m = lane%16
    const float* qp = Q + ((size_t)(b * SEQ + qrow) * HEADS + h) * DH;
#pragma unroll
    for (int c = 0; c < 2; ++c) {        // k-chunks d in [32c, 32c+32)
      AF a;
#pragma unroll
      for (int r = 0; r < 2; ++r) {      // two runs of 8 contiguous d
        const float* src = qp + 32 * c + 16 * r + 8 * hi;
        const v4f lo = *(const v4f*)(src);
        const v4f hi4 = *(const v4f*)(src + 4);
#pragma unroll
        for (int j = 0; j < 4; ++j) {
          a.e[8 * r + j]     = (f16)lo[j];
          a.e[8 * r + 4 + j] = (f16)hi4[j];
        }
      }
      qa[c] = a.v;
    }
  }

  v8f acc[4];                            // output tile 16x64 (4 d-chunks of 16)
#pragma unroll
  for (int dc = 0; dc < 4; ++dc) acc[dc] = (v8f){};
  float mrow[8], lrow[8];                // running row max (log2 domain), sum
#pragma unroll
  for (int j = 0; j < 8; ++j) { mrow[j] = -__builtin_inff(); lrow[j] = 0.0f; }

  for (int kb = 0; kb < SEQ; kb += BK) {
    __syncthreads();                     // protect LDS from previous iter reads
    // ---- cooperative K/V chunk load (64x64 each), float4, fp32 -> f16 ----
#pragma unroll
    for (int i = 0; i < 8; ++i) {
      const int e  = (tid + 128 * i) << 2;  // element 0..4092, step 4
      const int kk = e >> 6;                // key row in chunk (0..63)
      const int dd = e & 63;                // d (multiple of 4)
      const size_t g = ((size_t)(b * SEQ + kb + kk) * HEADS + h) * DH + dd;
      const v4f kv = *(const v4f*)(K + g);
      const v4f vv = *(const v4f*)(V + g);
      v4h kh;
#pragma unroll
      for (int j = 0; j < 4; ++j) kh[j] = (f16)kv[j];
      *(v4h*)(&kbuf[kk * KSTR + dd]) = kh;     // contiguous b64 store
#pragma unroll
      for (int j = 0; j < 4; ++j)
        vbufT[(dd + j) * VSTR + kk] = (f16)vv[j];  // transposed scatter
    }
    __syncthreads();

    // ---- scores: S2 = Q*K^T * (scale*log2e) + mask*log2e, 4 N-tiles ----
    v8f sc[4];
#pragma unroll
    for (int t = 0; t < 4; ++t) {
      const int key = 16 * t + n16;      // lane holds column n = key
      const f16* krow = &kbuf[key * KSTR];
      AF kb0, kb1;
      *(v8h*)&kb0.e[0] = *(const v8h*)(krow + 8 * hi);        // d 0..31 slice
      *(v8h*)&kb0.e[8] = *(const v8h*)(krow + 16 + 8 * hi);
      *(v8h*)&kb1.e[0] = *(const v8h*)(krow + 32 + 8 * hi);   // d 32..63 slice
      *(v8h*)&kb1.e[8] = *(const v8h*)(krow + 48 + 8 * hi);
      const v16h b0 = kb0.v;
      const v16h b1 = kb1.v;
      v8f s = (v8f){};
      s = wmma_f16(qa[0], b0, s);
      s = wmma_f16(qa[1], b1, s);
      const float mv2 = Mask[(size_t)b * SEQ + kb + key] * LOG2E;
#pragma unroll
      for (int j = 0; j < 8; ++j) s[j] = s[j] * kscale + mv2;
      sc[t] = s;
    }

    // ---- online softmax in log2 domain (row = 16 lanes, DPP butterfly) ----
    float alpha[8];
#pragma unroll
    for (int j = 0; j < 8; ++j) {
      const float a01 = fmaxf(sc[0][j], sc[1][j]);
      const float a23 = fmaxf(sc[2][j], sc[3][j]);
      const float mx = redmax16(fmaxf(a01, a23));
      const float mnew = fmaxf(mrow[j], mx);
      alpha[j] = __builtin_exp2f(mrow[j] - mnew);
      mrow[j] = mnew;
      const float p0 = __builtin_exp2f(sc[0][j] - mnew);
      const float p1 = __builtin_exp2f(sc[1][j] - mnew);
      const float p2 = __builtin_exp2f(sc[2][j] - mnew);
      const float p3 = __builtin_exp2f(sc[3][j] - mnew);
      sc[0][j] = p0; sc[1][j] = p1; sc[2][j] = p2; sc[3][j] = p3;
      const float rs = redsum16((p0 + p1) + (p2 + p3));
      lrow[j] = lrow[j] * alpha[j] + rs;
    }
#pragma unroll
    for (int dc = 0; dc < 4; ++dc)
#pragma unroll
      for (int j = 0; j < 8; ++j) acc[dc][j] *= alpha[j];

    // ---- stage P through per-wave LDS: C-layout -> A-layout, f32 -> f16 ----
    f16* pb = &pbuf[wave * 16 * PSTR];
#pragma unroll
    for (int t = 0; t < 4; ++t)
#pragma unroll
      for (int j = 0; j < 8; ++j)
        pb[(j + 8 * hi) * PSTR + 16 * t + n16] = (f16)sc[t][j];
    asm volatile("s_wait_dscnt 0x0" ::: "memory"); // same-wave LDS RAW fence

    v16h pa[2];
    const f16* prow = pb + n16 * PSTR;   // lane holds row m = lane%16
#pragma unroll
    for (int kc = 0; kc < 2; ++kc) {     // key-chunks 0..31, 32..63
      AF a;
      *(v8h*)&a.e[0] = *(const v8h*)(prow + 32 * kc + 8 * hi);
      *(v8h*)&a.e[8] = *(const v8h*)(prow + 32 * kc + 16 + 8 * hi);
      pa[kc] = a.v;
    }

    // ---- O += P * V  (k = 64 keys, 4 d-chunks of 16) ----
#pragma unroll
    for (int dc = 0; dc < 4; ++dc) {
      const f16* vcol = &vbufT[(16 * dc + n16) * VSTR];    // lane holds col d
      AF va0, va1;
      *(v8h*)&va0.e[0] = *(const v8h*)(vcol + 8 * hi);       // keys 0..31
      *(v8h*)&va0.e[8] = *(const v8h*)(vcol + 16 + 8 * hi);
      *(v8h*)&va1.e[0] = *(const v8h*)(vcol + 32 + 8 * hi);  // keys 32..63
      *(v8h*)&va1.e[8] = *(const v8h*)(vcol + 48 + 8 * hi);
      const v16h vb0 = va0.v;
      const v16h vb1 = va1.v;
      acc[dc] = wmma_f16(pa[0], vb0, acc[dc]);
      acc[dc] = wmma_f16(pa[1], vb1, acc[dc]);
    }
  }

  // ---- epilogue: normalize and store [B,H,S,D] fp32 ----
#pragma unroll
  for (int j = 0; j < 8; ++j) {
    const float r = 1.0f / lrow[j];
    const int q = q0 + j + 8 * hi;
    const size_t ob = ((size_t)bh * SEQ + q) * DH;
#pragma unroll
    for (int dc = 0; dc < 4; ++dc)
      Out[ob + 16 * dc + n16] = acc[dc][j] * r;
  }
}

extern "C" void kernel_launch(void* const* d_in, const int* in_sizes, int n_in,
                              void* d_out, int out_size, void* d_ws, size_t ws_size,
                              hipStream_t stream) {
  (void)in_sizes; (void)n_in; (void)out_size; (void)d_ws; (void)ws_size;
  const float* Q    = (const float*)d_in[0];
  const float* K    = (const float*)d_in[1];
  const float* V    = (const float*)d_in[2];
  const float* Mask = (const float*)d_in[3];
  const int*   IS   = (const int*)d_in[4];
  float* Out = (float*)d_out;
  dim3 grid(SEQ / BQ, BATCH * HEADS);    // (32, 32)
  dim3 block(128);
  flash_attn_f16_wmma<<<grid, block, 0, stream>>>(Q, K, V, Mask, IS, Out);
}
